// SelfAttentionModule_11201274708186
// MI455X (gfx1250) — matrix operand order repaired and checked by reading, not compiled, not run on hardware
//
#include <hip/hip_runtime.h>
#include <hip/hip_bf16.h>

// ---------------- CDNA5 WMMA types ----------------
typedef __attribute__((ext_vector_type(16))) __bf16 v16b;
typedef __attribute__((ext_vector_type(8)))  float  v8f;

union B16Frag { uint4 q[2]; unsigned short us[16]; v16b v; };

static __device__ __forceinline__ v8f wmma_bf16(v16b a, v16b b, v8f c) {
  return __builtin_amdgcn_wmma_f32_16x16x32_bf16(false, a, false, b, (short)0, c, false, false);
}

static __device__ __forceinline__ unsigned short f2bf(float f) {
  union { float f; unsigned int u; } a; a.f = f;
  unsigned int r = a.u + 0x7FFFu + ((a.u >> 16) & 1u);   // RNE
  return (unsigned short)(r >> 16);
}

// ---- CDNA5 async global->LDS copy (ASYNCcnt path, bypasses VGPRs) ----
// GLOBAL_LOAD_ASYNC_TO_LDS_B128: VDST = LDS byte address (low 32 bits of the
// generic shared pointer), VADDR = 64-bit global address, GV mode (saddr=off).
static __device__ __forceinline__ void async_copy16(void* ldsDst, const void* gsrc) {
  unsigned int l = (unsigned int)(unsigned long long)ldsDst;
  asm volatile("global_load_async_to_lds_b128 %0, %1, off"
               :: "v"(l), "v"(gsrc)
               : "memory");
}
static __device__ __forceinline__ void wait_async0() {
  asm volatile("s_wait_asynccnt 0x0" ::: "memory");
}

// ---------------- problem constants ----------------
#define BB   4
#define LL   1024
#define DD   1024
#define HH   16
#define DHH  64
#define MM2  2047          // 2L-1
#define BLR  4096          // B*L rows
#define PMR  8188          // B*(2L-1) rows
#define PMRP 8192          // padded rows for unguarded async staging

// ---------------- workspace layout (bytes) ----------------
static const size_t OFF_XN  = 0;                                   // [BLR,D] bf16
static const size_t OFF_WQT = OFF_XN  + (size_t)BLR*DD*2;          // [D,D] bf16 (transposed)
static const size_t OFF_WKT = OFF_WQT + (size_t)DD*DD*2;
static const size_t OFF_WVT = OFF_WKT + (size_t)DD*DD*2;
static const size_t OFF_WPT = OFF_WVT + (size_t)DD*DD*2;
static const size_t OFF_WOT = OFF_WPT + (size_t)DD*DD*2;
static const size_t OFF_POS = OFF_WOT + (size_t)DD*DD*2;           // [PMRP,D] bf16 (padded)
static const size_t OFF_QU  = OFF_POS + (size_t)PMRP*DD*2;         // [BH,L,64] bf16
static const size_t OFF_QV  = OFF_QU  + (size_t)BLR*DD*2;
static const size_t OFF_KK  = OFF_QV  + (size_t)BLR*DD*2;          // [BH,L,64]
static const size_t OFF_VT  = OFF_KK  + (size_t)BLR*DD*2;          // [BH,64,L]
static const size_t OFF_PB  = OFF_VT  + (size_t)BLR*DD*2;          // [BH,2048,64]
static const size_t OFF_OB  = OFF_PB  + (size_t)BB*HH*2048*DHH*2;  // [BLR,D] bf16

// d_out sections (float elements): out, attn, content_score, pos_score
static const size_t DO_ATT = (size_t)BB*LL*DD;                     // 4,194,304
static const size_t DO_CSC = DO_ATT + (size_t)BB*HH*LL*LL;         // +67,108,864
static const size_t DO_PSC = DO_CSC + (size_t)BB*HH*LL*LL;

// ================= LayerNorm -> bf16 =================
__global__ __launch_bounds__(256)
void ln_kernel(const float* __restrict__ x, const float* __restrict__ g,
               const float* __restrict__ b, unsigned short* __restrict__ xn) {
  __shared__ float red[256];
  int row = blockIdx.x, t = threadIdx.x;
  const float4 v  = ((const float4*)(x + (size_t)row*DD))[t];
  float s = v.x + v.y + v.z + v.w;
  red[t] = s; __syncthreads();
  for (int o = 128; o > 0; o >>= 1) { if (t < o) red[t] += red[t+o]; __syncthreads(); }
  float mu = red[0] * (1.0f/DD); __syncthreads();
  float dx0=v.x-mu, dx1=v.y-mu, dx2=v.z-mu, dx3=v.w-mu;
  red[t] = dx0*dx0 + dx1*dx1 + dx2*dx2 + dx3*dx3; __syncthreads();
  for (int o = 128; o > 0; o >>= 1) { if (t < o) red[t] += red[t+o]; __syncthreads(); }
  float rstd = rsqrtf(red[0] * (1.0f/DD) + 1e-12f);
  const float4 gv = ((const float4*)g)[t];
  const float4 bv = ((const float4*)b)[t];
  unsigned short* o = xn + (size_t)row*DD + t*4;
  o[0] = f2bf(dx0*rstd*gv.x + bv.x);
  o[1] = f2bf(dx1*rstd*gv.y + bv.y);
  o[2] = f2bf(dx2*rstd*gv.z + bv.z);
  o[3] = f2bf(dx3*rstd*gv.w + bv.w);
}

// ============ transpose+convert fp32 [K][N] -> bf16 [N][K] ============
__global__ __launch_bounds__(256)
void tcvt_kernel(const float* __restrict__ in, unsigned short* __restrict__ out,
                 int rows, int cols) {
  __shared__ float tile[32][33];
  int x = blockIdx.x*32 + threadIdx.x;
  int y0 = blockIdx.y*32;
  for (int j = threadIdx.y; j < 32; j += 8)
    tile[j][threadIdx.x] = in[(size_t)(y0 + j)*cols + x];
  __syncthreads();
  int ox  = blockIdx.y*32 + threadIdx.x;   // out col = original row (K)
  int oy0 = blockIdx.x*32;                 // out row = original col (N)
  for (int j = threadIdx.y; j < 32; j += 8)
    out[(size_t)(oy0 + j)*rows + ox] = f2bf(tile[threadIdx.x][j]);
}

// ============ elementwise fp32 -> bf16 (x4) ============
__global__ __launch_bounds__(256)
void cvt_kernel(const float* __restrict__ in, unsigned short* __restrict__ out, long n4) {
  long i = (long)blockIdx.x*256 + threadIdx.x;
  if (i < n4) {
    float4 f = ((const float4*)in)[i];
    ushort4 o; o.x=f2bf(f.x); o.y=f2bf(f.y); o.z=f2bf(f.z); o.w=f2bf(f.w);
    ((ushort4*)out)[i] = o;
  }
}

// zero the pad row (2047) of p [BH,2048,64]
__global__ void zero_ppad_kernel(unsigned short* __restrict__ pp) {
  pp[(((size_t)blockIdx.x*2048) + 2047)*DHH + threadIdx.x] = 0;
}

// ================= projection GEMM (bf16 WMMA, async-LDS staging) =================
#define MODE_Q 0
#define MODE_K 1
#define MODE_V 2
#define MODE_P 3
#define MODE_O 4
#define BMT 128
#define BNT 128
#define BKT 32

__global__ __launch_bounds__(256)
void gemm_proj_kernel(const unsigned short* __restrict__ A,    // [Mp,1024] bf16 (padded alloc)
                      const unsigned short* __restrict__ Bt,   // [1024,1024] bf16 [N][K]
                      const float* __restrict__ bias,          // [N] or null
                      const float* __restrict__ ub,            // u (MODE_Q)
                      const float* __restrict__ vb,            // v (MODE_Q)
                      unsigned short* __restrict__ out0,
                      unsigned short* __restrict__ out1,       // qv (MODE_Q)
                      float* __restrict__ outf,                // MODE_O
                      int M, int mode) {
  const int K = 1024;
  __shared__ unsigned short As[2][BMT][BKT];
  __shared__ unsigned short Bs[2][BNT][BKT];
  const int tid  = threadIdx.x;
  const int lane = tid & 31, wave = tid >> 5;
  const int wm = wave & 3, wn = wave >> 2;
  const int g = lane >> 4, r = lane & 15;
  const int blockM = blockIdx.y * BMT, blockN = blockIdx.x * BNT;
  const int srow = tid >> 1, scol = (tid & 1) * 16;

  v8f acc[2][4];
  for (int i = 0; i < 2; i++) for (int j = 0; j < 4; j++)
    for (int e = 0; e < 8; e++) acc[i][j][e] = 0.0f;

  // unguarded async staging: A allocation is padded past M (MODE_P), and
  // out-of-range rows only pollute output rows dropped by the epilogue guard.
  auto stageA = [&](int buf, int kb) {
    const unsigned short* src = A + (size_t)(blockM + srow)*K + kb + scol;
    async_copy16(&As[buf][srow][scol],     src);
    async_copy16(&As[buf][srow][scol + 8], src + 8);
  };
  auto stageB = [&](int buf, int kb) {
    const unsigned short* src = Bt + (size_t)(blockN + srow)*K + kb + scol;
    async_copy16(&Bs[buf][srow][scol],     src);
    async_copy16(&Bs[buf][srow][scol + 8], src + 8);
  };

  stageA(0, 0); stageB(0, 0);
  wait_async0();
  __syncthreads();
  int buf = 0;
  for (int kb = 0; kb < K; kb += BKT, buf ^= 1) {
    if (kb + BKT < K) { stageA(buf ^ 1, kb + BKT); stageB(buf ^ 1, kb + BKT); }
    if (kb + 2*BKT < K) {  // speculative prefetch of K+2 tile -> global_prefetch_b8
      __builtin_prefetch(A  + (size_t)(blockM + srow)*K + kb + 2*BKT + scol, 0, 1);
      __builtin_prefetch(Bt + (size_t)(blockN + srow)*K + kb + 2*BKT + scol, 0, 1);
    }
    B16Frag af[2], bfr[4];
    for (int mi = 0; mi < 2; mi++) {
      const unsigned short* p = &As[buf][wm*32 + mi*16 + r][g*8];
      af[mi].q[0] = *(const uint4*)p;
      af[mi].q[1] = *(const uint4*)(p + 16);
    }
    for (int ni = 0; ni < 4; ni++) {
      const unsigned short* p = &Bs[buf][wn*64 + ni*16 + r][g*8];
      bfr[ni].q[0] = *(const uint4*)p;
      bfr[ni].q[1] = *(const uint4*)(p + 16);
    }
    for (int mi = 0; mi < 2; mi++)
      for (int ni = 0; ni < 4; ni++)
        acc[mi][ni] = wmma_bf16(af[mi].v, bfr[ni].v, acc[mi][ni]);
    wait_async0();
    __syncthreads();
  }

  for (int mi = 0; mi < 2; mi++) {
    int mbase = blockM + wm*32 + mi*16 + g*8;
    for (int ni = 0; ni < 4; ni++) {
      int n = blockN + wn*64 + ni*16 + r;
      float badd = bias ? bias[n] : 0.0f;
      int h = n >> 6, d = n & 63;
      float ua = 0.f, va = 0.f;
      if (mode == MODE_Q) { ua = ub[h*DHH + d]; va = vb[h*DHH + d]; }
      for (int e = 0; e < 8; e++) {
        int m = mbase + e;
        float f = acc[mi][ni][e] + badd;
        if (mode == MODE_Q) {
          int bb = m >> 10, l = m & 1023;
          size_t idx = (((size_t)bb*HH + h)*LL + l)*DHH + d;
          out0[idx] = f2bf(f + ua);
          out1[idx] = f2bf(f + va);
        } else if (mode == MODE_K) {
          int bb = m >> 10, l = m & 1023;
          out0[(((size_t)bb*HH + h)*LL + l)*DHH + d] = f2bf(f);
        } else if (mode == MODE_V) {
          int bb = m >> 10, l = m & 1023;
          out0[(((size_t)bb*HH + h)*DHH + d)*LL + l] = f2bf(f);   // vT [BH,64,L]
        } else if (mode == MODE_P) {
          if (m < M) {
            int bb = m / MM2, rr = m - bb*MM2;
            out0[(((size_t)bb*HH + h)*2048 + rr)*DHH + d] = f2bf(f);
          }
        } else { // MODE_O
          outf[(size_t)m*DD + n] = f;
        }
      }
    }
  }
}

// ================= scores: content + shifted positional =================
__global__ __launch_bounds__(256)
void scores_kernel(const unsigned short* __restrict__ qu,  // [BH,L,64]
                   const unsigned short* __restrict__ qv,
                   const unsigned short* __restrict__ kk,  // [BH,L,64]
                   const unsigned short* __restrict__ pp,  // [BH,2048,64]
                   float* __restrict__ csc, float* __restrict__ psc) {
  const int bh = blockIdx.z;
  const int tid = threadIdx.x, lane = tid & 31, wave = tid >> 5;
  const int g = lane >> 4, r = lane & 15, g16 = lane & 16;
  const int l0 = blockIdx.y*128 + wave*16;
  const unsigned short* qub = qu + (size_t)bh*LL*DHH;
  const unsigned short* qvb = qv + (size_t)bh*LL*DHH;
  const unsigned short* kbp = kk + (size_t)bh*LL*DHH;
  const unsigned short* pbp = pp + (size_t)bh*2048*DHH;

  B16Frag aq[2], av[2];
  for (int ks = 0; ks < 2; ks++) {
    const unsigned short* p = qub + (size_t)(l0 + r)*DHH + ks*32 + g*16;
    aq[ks].q[0] = *(const uint4*)p; aq[ks].q[1] = *(const uint4*)(p + 8);
    p = qvb + (size_t)(l0 + r)*DHH + ks*32 + g*16;
    av[ks].q[0] = *(const uint4*)p; av[ks].q[1] = *(const uint4*)(p + 8);
  }

  for (int mt = 0; mt < 8; mt++) {
    int m0 = blockIdx.x*128 + mt*16;
    v8f accC, t0, t1;
    for (int e = 0; e < 8; e++) { accC[e] = 0.f; t0[e] = 0.f; t1[e] = 0.f; }
    // content: (q+u) @ k^T
    for (int ks = 0; ks < 2; ks++) {
      B16Frag bk;
      const unsigned short* p = kbp + (size_t)(m0 + r)*DHH + ks*32 + g*16;
      bk.q[0] = *(const uint4*)p; bk.q[1] = *(const uint4*)(p + 8);
      accC = wmma_bf16(aq[ks].v, bk.v, accC);
    }
    // positional: strip over p rows [pbase, pbase+31], then diagonal extract
    int pbase = (LL - 1) + m0 - l0 - 15;   // >= 0, pbase+31 <= 2047
    for (int ks = 0; ks < 2; ks++) {
      B16Frag b0, b1;
      const unsigned short* p0 = pbp + (size_t)(pbase + r)*DHH + ks*32 + g*16;
      const unsigned short* p1 = pbp + (size_t)(pbase + 16 + r)*DHH + ks*32 + g*16;
      b0.q[0] = *(const uint4*)p0; b0.q[1] = *(const uint4*)(p0 + 8);
      b1.q[0] = *(const uint4*)p1; b1.q[1] = *(const uint4*)(p1 + 8);
      t0 = wmma_bf16(av[ks].v, b0.v, t0);
      t1 = wmma_bf16(av[ks].v, b1.v, t1);
    }
    for (int e = 0; e < 8; e++) {
      int row = l0 + e + g*8, col = m0 + r;
      size_t oidx = ((size_t)bh*LL + row)*LL + col;
      csc[oidx] = accC[e];
      // pos_score[row,col] = strip[row, col-row+15] relative to pbase
      int cp = r + (g16 ? 7 : 15) - e;               // 0..31
      int sl = g16 | (cp & 15);
      float s0 = __int_as_float(__builtin_amdgcn_ds_bpermute(sl << 2, __float_as_int(t0[e])));
      float s1 = __int_as_float(__builtin_amdgcn_ds_bpermute(sl << 2, __float_as_int(t1[e])));
      psc[oidx] = (cp < 16) ? s0 : s1;
    }
  }
}

// ================= softmax over (content+pos)/sqrt(dh) =================
__global__ __launch_bounds__(256)
void softmax_kernel(const float* __restrict__ csc, const float* __restrict__ psc,
                    float* __restrict__ attn) {
  __shared__ float red[256];
  size_t row = blockIdx.x;
  int t = threadIdx.x;
  const float4 c = ((const float4*)(csc + row*LL))[t];
  const float4 p = ((const float4*)(psc + row*LL))[t];
  float s0 = (c.x + p.x)*0.125f, s1 = (c.y + p.y)*0.125f;
  float s2 = (c.z + p.z)*0.125f, s3 = (c.w + p.w)*0.125f;
  float m = fmaxf(fmaxf(s0, s1), fmaxf(s2, s3));
  red[t] = m; __syncthreads();
  for (int o = 128; o > 0; o >>= 1) { if (t < o) red[t] = fmaxf(red[t], red[t+o]); __syncthreads(); }
  float mx = red[0]; __syncthreads();
  float e0 = expf(s0 - mx), e1 = expf(s1 - mx), e2 = expf(s2 - mx), e3 = expf(s3 - mx);
  red[t] = e0 + e1 + e2 + e3; __syncthreads();
  for (int o = 128; o > 0; o >>= 1) { if (t < o) red[t] += red[t+o]; __syncthreads(); }
  float inv = 1.0f / red[0];
  float4 o4; o4.x = e0*inv; o4.y = e1*inv; o4.z = e2*inv; o4.w = e3*inv;
  ((float4*)(attn + row*LL))[t] = o4;
}

// ================= attn @ v (fp32 A converted inline, async-LDS B) =================
__global__ __launch_bounds__(256)
void av_kernel(const float* __restrict__ attn,          // [BH,L,L] fp32
               const unsigned short* __restrict__ vT,   // [BH,64,L] bf16
               unsigned short* __restrict__ ob) {       // [BLR,D] bf16
  __shared__ unsigned short Bs[2][64][32];
  const int bh = blockIdx.y, b = bh >> 4, h = bh & 15;
  const int tid = threadIdx.x, lane = tid & 31, wave = tid >> 5;
  const int g = lane >> 4, r = lane & 15;
  const int l0 = blockIdx.x*128 + wave*16;
  const float* ab = attn + (size_t)bh*LL*LL;
  const unsigned short* vb = vT + (size_t)bh*DHH*LL;

  auto stage = [&](int buf, int kb) {
    int d = tid >> 2, part = tid & 3;
    async_copy16(&Bs[buf][d][part*8], vb + (size_t)d*LL + kb + part*8);
  };
  v8f acc[4];
  for (int j = 0; j < 4; j++) for (int e = 0; e < 8; e++) acc[j][e] = 0.f;

  stage(0, 0);
  wait_async0();
  __syncthreads();
  int buf = 0;
  for (int kb = 0; kb < LL; kb += 32, buf ^= 1) {
    if (kb + 32 < LL) stage(buf ^ 1, kb + 32);
    B16Frag af;
    const float* ap = ab + (size_t)(l0 + r)*LL + kb + g*8;
    float4 f0 = *(const float4*)ap;
    float4 f1 = *(const float4*)(ap + 4);
    float4 f2 = *(const float4*)(ap + 16);
    float4 f3 = *(const float4*)(ap + 20);
    af.us[0]=f2bf(f0.x);  af.us[1]=f2bf(f0.y);  af.us[2]=f2bf(f0.z);  af.us[3]=f2bf(f0.w);
    af.us[4]=f2bf(f1.x);  af.us[5]=f2bf(f1.y);  af.us[6]=f2bf(f1.z);  af.us[7]=f2bf(f1.w);
    af.us[8]=f2bf(f2.x);  af.us[9]=f2bf(f2.y);  af.us[10]=f2bf(f2.z); af.us[11]=f2bf(f2.w);
    af.us[12]=f2bf(f3.x); af.us[13]=f2bf(f3.y); af.us[14]=f2bf(f3.z); af.us[15]=f2bf(f3.w);
    for (int ni = 0; ni < 4; ni++) {
      B16Frag bfr;
      const unsigned short* p = &Bs[buf][ni*16 + r][g*8];
      bfr.q[0] = *(const uint4*)p; bfr.q[1] = *(const uint4*)(p + 16);
      acc[ni] = wmma_bf16(af.v, bfr.v, acc[ni]);
    }
    wait_async0();
    __syncthreads();
  }
  for (int ni = 0; ni < 4; ni++) {
    int n = h*DHH + ni*16 + r;
    for (int e = 0; e < 8; e++) {
      int row = l0 + e + g*8;
      ob[((size_t)(b*LL + row))*DD + n] = f2bf(acc[ni][e]);
    }
  }
}

// ================= host launch =================
extern "C" void kernel_launch(void* const* d_in, const int* in_sizes, int n_in,
                              void* d_out, int out_size, void* d_ws, size_t ws_size,
                              hipStream_t stream) {
  const float* x   = (const float*)d_in[0];
  const float* pe  = (const float*)d_in[1];
  const float* lng = (const float*)d_in[2];
  const float* lnb = (const float*)d_in[3];
  const float* Wq  = (const float*)d_in[4];
  const float* bq  = (const float*)d_in[5];
  const float* Wk  = (const float*)d_in[6];
  const float* bk  = (const float*)d_in[7];
  const float* Wv  = (const float*)d_in[8];
  const float* bv  = (const float*)d_in[9];
  const float* Wp  = (const float*)d_in[10];
  const float* Wo  = (const float*)d_in[11];
  const float* bo  = (const float*)d_in[12];
  const float* u   = (const float*)d_in[13];
  const float* v   = (const float*)d_in[14];

  char* ws = (char*)d_ws;
  unsigned short* xn  = (unsigned short*)(ws + OFF_XN);
  unsigned short* wqT = (unsigned short*)(ws + OFF_WQT);
  unsigned short* wkT = (unsigned short*)(ws + OFF_WKT);
  unsigned short* wvT = (unsigned short*)(ws + OFF_WVT);
  unsigned short* wpT = (unsigned short*)(ws + OFF_WPT);
  unsigned short* woT = (unsigned short*)(ws + OFF_WOT);
  unsigned short* pos = (unsigned short*)(ws + OFF_POS);
  unsigned short* qu  = (unsigned short*)(ws + OFF_QU);
  unsigned short* qv  = (unsigned short*)(ws + OFF_QV);
  unsigned short* kk  = (unsigned short*)(ws + OFF_KK);
  unsigned short* vT  = (unsigned short*)(ws + OFF_VT);
  unsigned short* pb  = (unsigned short*)(ws + OFF_PB);
  unsigned short* ob  = (unsigned short*)(ws + OFF_OB);

  float* outp = (float*)d_out;
  float* attn = outp + DO_ATT;
  float* csc  = outp + DO_CSC;
  float* psc  = outp + DO_PSC;

  // 1. LayerNorm + bf16
  ln_kernel<<<BLR, 256, 0, stream>>>(x, lng, lnb, xn);
  // 2. weights -> bf16 transposed [N][K]; pos_emb -> bf16
  dim3 tb(32, 8), tg(32, 32);
  tcvt_kernel<<<tg, tb, 0, stream>>>(Wq, wqT, DD, DD);
  tcvt_kernel<<<tg, tb, 0, stream>>>(Wk, wkT, DD, DD);
  tcvt_kernel<<<tg, tb, 0, stream>>>(Wv, wvT, DD, DD);
  tcvt_kernel<<<tg, tb, 0, stream>>>(Wp, wpT, DD, DD);
  tcvt_kernel<<<tg, tb, 0, stream>>>(Wo, woT, DD, DD);
  cvt_kernel<<<(PMR*DD/4 + 255)/256, 256, 0, stream>>>(pe, pos, (long)PMR*DD/4);
  zero_ppad_kernel<<<BB*HH, DHH, 0, stream>>>(pb);
  // 3. projections (bf16 WMMA GEMMs, async-LDS staging)
  gemm_proj_kernel<<<dim3(8, 32), 256, 0, stream>>>(xn, wqT, bq, u, v, qu, qv, nullptr, BLR, MODE_Q);
  gemm_proj_kernel<<<dim3(8, 32), 256, 0, stream>>>(xn, wkT, bk, nullptr, nullptr, kk, nullptr, nullptr, BLR, MODE_K);
  gemm_proj_kernel<<<dim3(8, 32), 256, 0, stream>>>(xn, wvT, bv, nullptr, nullptr, vT, nullptr, nullptr, BLR, MODE_V);
  gemm_proj_kernel<<<dim3(8, 64), 256, 0, stream>>>(pos, wpT, nullptr, nullptr, nullptr, pb, nullptr, nullptr, PMR, MODE_P);
  // 4. content + shifted positional scores (direct, no rel_shift buffer)
  scores_kernel<<<dim3(8, 8, BB*HH), 256, 0, stream>>>(qu, qv, kk, pb, csc, psc);
  // 5. softmax
  softmax_kernel<<<BB*HH*LL, 256, 0, stream>>>(csc, psc, attn);
  // 6. attn @ v
  av_kernel<<<dim3(8, BB*HH), 256, 0, stream>>>(attn, vT, ob);
  // 7. output projection
  gemm_proj_kernel<<<dim3(8, 32), 256, 0, stream>>>(ob, woT, bo, nullptr, nullptr, nullptr, nullptr, outp, BLR, MODE_O);
}
